// Attention_55783035240731
// MI455X (gfx1250) — matrix operand order repaired and checked by reading, not compiled, not run on hardware
//
#include <hip/hip_runtime.h>

typedef __attribute__((ext_vector_type(16))) _Float16 v16h;
typedef __attribute__((ext_vector_type(8)))  float    v8f;

constexpr int B_ = 4, S_ = 4096, E_ = 512, D_ = 64;
constexpr int ROWS = B_ * S_;                       // 16384
constexpr float SCALE = 0.04419417382415922f;       // 1/sqrt(512)  (reference uses E!)

// ---------------------------------------------------------------------------
// Kernel 1: convert WQ/WK/WV (fp32 [E,D]) to f16, pre-swizzled into WMMA
// B-fragment lane order: packed[w][ec][nt][lane][h] = W[ec*32+(lane>>4)*16+h][nt*16+(lane&15)]
// so each lane's fragment is one contiguous v16h (32B) load.
// ---------------------------------------------------------------------------
__global__ __launch_bounds__(256) void prep_pack_w(
    const float* __restrict__ WQ, const float* __restrict__ WK,
    const float* __restrict__ WV, _Float16* __restrict__ packed)
{
    int tid = blockIdx.x * 256 + threadIdx.x;
    if (tid >= 3 * E_ * D_) return;
    int w      = tid / (E_ * D_);
    int r      = tid - w * (E_ * D_);
    int frag   = r >> 9;          // 512 halves per fragment
    int within = r & 511;
    int lane   = within >> 4;
    int h      = within & 15;
    int ec     = frag >> 2;
    int nt     = frag & 3;
    int e      = ec * 32 + ((lane >> 4) * 16) + h;
    int d      = nt * 16 + (lane & 15);
    const float* W = (w == 0) ? WQ : ((w == 1) ? WK : WV);
    packed[tid] = (_Float16)W[e * D_ + d];
}

// ---------------------------------------------------------------------------
// Kernel 2: QKV projection. One wave per 16-row tile of x.
// Q, K stored row-major f16 [ROWS, D] (row-major K == contiguous B fragments
// for the scores GEMM). V stored TRANSPOSED f16 [B][D][S] so the P@V
// B fragment (16 contiguous keys per lane at fixed d) is one v16h load.
// ---------------------------------------------------------------------------
__global__ __launch_bounds__(256) void qkv_proj(
    const float* __restrict__ x, const _Float16* __restrict__ packedW,
    _Float16* __restrict__ Qo, _Float16* __restrict__ Ko, _Float16* __restrict__ Vt)
{
    const int lane = threadIdx.x & 31;
    const int wv   = threadIdx.x >> 5;
    const int rt   = blockIdx.x * 8 + wv;     // 16-row tile index
    const int r0   = rt * 16;
    const int half = lane >> 4;
    const int l15  = lane & 15;
    const int abase = half * 16;

    v8f acc[3][4] = {};
    const float* xrow = x + (size_t)(r0 + l15) * E_;

    for (int ec = 0; ec < 16; ++ec) {
        v16h a;
        {
            const float* p0 = xrow + ec * 32 + abase;
#pragma unroll
            for (int k = 0; k < 8; ++k) {
                a[k]     = (_Float16)p0[k];
                a[8 + k] = (_Float16)p0[8 + k];
            }
        }
#pragma unroll
        for (int w = 0; w < 3; ++w) {
#pragma unroll
            for (int nt = 0; nt < 4; ++nt) {
                const _Float16* bp =
                    packedW + (size_t)w * (E_ * D_) + (((ec * 4 + nt) * 32 + lane) * 16);
                v16h b = *(const v16h*)bp;
                acc[w][nt] = __builtin_amdgcn_wmma_f32_16x16x32_f16(
                    false, a, false, b, (short)0, acc[w][nt], false, false);
            }
        }
    }
    // C layout: lane holds column n=l15, rows m = i + 8*half.
    // Q, K: row-major stores.
#pragma unroll
    for (int w = 0; w < 2; ++w) {
        _Float16* O = (w == 0) ? Qo : Ko;
#pragma unroll
        for (int nt = 0; nt < 4; ++nt) {
#pragma unroll
            for (int i = 0; i < 8; ++i) {
                int m = i + 8 * half;
                O[(size_t)(r0 + m) * D_ + nt * 16 + l15] = (_Float16)acc[w][nt][i];
            }
        }
    }
    // V: transposed store Vt[b][d][s]  (each 16-row tile lies in one batch)
    {
        const int b = r0 / S_;
        const int s0 = r0 - b * S_;
#pragma unroll
        for (int nt = 0; nt < 4; ++nt) {
            const int d = nt * 16 + l15;
            _Float16* vt = Vt + ((size_t)b * D_ + d) * S_ + s0;
#pragma unroll
            for (int i = 0; i < 8; ++i) {
                int m = i + 8 * half;
                vt[m] = (_Float16)acc[2][nt][i];
            }
        }
    }
}

// ---------------------------------------------------------------------------
// Kernel 3: flash attention. One wave per 16-query tile; 32-key blocks.
// ---------------------------------------------------------------------------
__global__ __launch_bounds__(128) void flash_attn(
    const _Float16* __restrict__ Q, const _Float16* __restrict__ K,
    const _Float16* __restrict__ Vt, float* __restrict__ out)
{
    __shared__ _Float16 Plds[4][16 * 32];   // per-wave P transpose tile (1KB each)

    const int lane = threadIdx.x & 31;
    const int wv   = threadIdx.x >> 5;
    const int tile = blockIdx.x * 4 + wv;
    const int b    = tile >> 8;             // 256 q-tiles per batch
    const int qt   = tile & 255;
    const int q0   = qt * 16;
    const int half = lane >> 4;
    const int l15  = lane & 15;
    const size_t rowbase = (size_t)b * S_;

    // Q A-fragments for d-chunks 0..31 and 32..63
    v16h aq[2];
    {
        const _Float16* qrow = Q + (rowbase + q0 + l15) * D_;
        const int base = half * 16;
#pragma unroll
        for (int c = 0; c < 2; ++c) {
            const _Float16* p0 = qrow + c * 32 + base;
#pragma unroll
            for (int k = 0; k < 8; ++k) { aq[c][k] = p0[k]; aq[c][8 + k] = p0[8 + k]; }
        }
    }

    v8f O[4] = {};
    float Mrow[8], Lrow[8];
#pragma unroll
    for (int i = 0; i < 8; ++i) { Mrow[i] = -3.0e38f; Lrow[i] = 0.0f; }

    // Per-lane V^T fragment base: lane reads 16 contiguous keys at fixed d
    const int kb = half * 16;

    const int qmax = q0 + 15;
    for (int k0 = 0; k0 <= qmax; k0 += 32) {
        // ---- scores: two 16-key C fragments, K rows read contiguously ----
        v8f c[2] = {};
#pragma unroll
        for (int f = 0; f < 2; ++f) {
            const _Float16* krow = K + (rowbase + k0 + f * 16 + l15) * D_;
#pragma unroll
            for (int dc = 0; dc < 2; ++dc) {
                v16h bk = *(const v16h*)(krow + dc * 32 + kb);
                c[f] = __builtin_amdgcn_wmma_f32_16x16x32_f16(
                    false, aq[dc], false, bk, (short)0, c[f], false, false);
            }
        }
        // ---- scale + causal mask ----
#pragma unroll
        for (int f = 0; f < 2; ++f) {
            const int kk = k0 + f * 16 + l15;
#pragma unroll
            for (int i = 0; i < 8; ++i) {
                const int q = q0 + i + 8 * half;
                float s = c[f][i] * SCALE;
                c[f][i] = (kk <= q) ? s : -1.0e30f;
            }
        }
        // ---- online softmax (row = VGPR index, reduce across 16-lane half) ----
#pragma unroll
        for (int i = 0; i < 8; ++i) {
            float bm = fmaxf(c[0][i], c[1][i]);
            bm = fmaxf(bm, __shfl_xor(bm, 1, 32));
            bm = fmaxf(bm, __shfl_xor(bm, 2, 32));
            bm = fmaxf(bm, __shfl_xor(bm, 4, 32));
            bm = fmaxf(bm, __shfl_xor(bm, 8, 32));
            float newM = fmaxf(Mrow[i], bm);
            float fac  = __expf(Mrow[i] - newM);
            float p0 = __expf(c[0][i] - newM);
            float p1 = __expf(c[1][i] - newM);
            c[0][i] = p0; c[1][i] = p1;
            float rs = p0 + p1;
            rs += __shfl_xor(rs, 1, 32);
            rs += __shfl_xor(rs, 2, 32);
            rs += __shfl_xor(rs, 4, 32);
            rs += __shfl_xor(rs, 8, 32);
            Lrow[i] = Lrow[i] * fac + rs;
            Mrow[i] = newM;
#pragma unroll
            for (int dt = 0; dt < 4; ++dt) O[dt][i] *= fac;
        }
        // ---- transpose P (C layout -> A layout) through wave-private LDS ----
        _Float16* P = &Plds[wv][0];
#pragma unroll
        for (int f = 0; f < 2; ++f) {
#pragma unroll
            for (int i = 0; i < 8; ++i) {
                const int m = i + 8 * half;
                P[m * 32 + f * 16 + l15] = (_Float16)c[f][i];
            }
        }
        // wave-local LDS RAW fence (no workgroup barrier: waves have different trip counts)
        asm volatile("s_wait_dscnt 0x0" ::: "memory");
        v16h ap;
        {
            const int base = half * 16;
            const _Float16* pm = P + l15 * 32;
#pragma unroll
            for (int k = 0; k < 8; ++k) { ap[k] = pm[base + k]; ap[8 + k] = pm[base + 8 + k]; }
        }
        // ---- O += P @ V : B fragment is one contiguous v16h from V^T ----
#pragma unroll
        for (int dt = 0; dt < 4; ++dt) {
            const int d = dt * 16 + l15;
            v16h bv = *(const v16h*)(Vt + ((size_t)b * D_ + d) * S_ + k0 + kb);
            O[dt] = __builtin_amdgcn_wmma_f32_16x16x32_f16(
                false, ap, false, bv, (short)0, O[dt], false, false);
        }
    }
    // ---- epilogue: out = O / L ----
#pragma unroll
    for (int dt = 0; dt < 4; ++dt) {
#pragma unroll
        for (int i = 0; i < 8; ++i) {
            const int m = i + 8 * half;
            out[(rowbase + q0 + m) * (size_t)D_ + dt * 16 + l15] = O[dt][i] / Lrow[i];
        }
    }
}

// ---------------------------------------------------------------------------
extern "C" void kernel_launch(void* const* d_in, const int* in_sizes, int n_in,
                              void* d_out, int out_size, void* d_ws, size_t ws_size,
                              hipStream_t stream) {
    (void)in_sizes; (void)n_in; (void)out_size; (void)ws_size;
    const float* x  = (const float*)d_in[0];
    const float* WQ = (const float*)d_in[1];
    const float* WK = (const float*)d_in[2];
    const float* WV = (const float*)d_in[3];
    float* out = (float*)d_out;

    char* ws = (char*)d_ws;
    _Float16* packedW = (_Float16*)ws;                                  // 196608 B
    _Float16* Qh = (_Float16*)(ws + 196608);                            // 2 MB
    _Float16* Kh = (_Float16*)(ws + 196608 + (size_t)ROWS * D_ * 2);    // 2 MB
    _Float16* Vh = (_Float16*)(ws + 196608 + 2 * (size_t)ROWS * D_ * 2);// 2 MB (V^T)

    prep_pack_w<<<(3 * E_ * D_ + 255) / 256, 256, 0, stream>>>(WQ, WK, WV, packedW);
    qkv_proj  <<<(ROWS / 16) / 8, 256, 0, stream>>>(x, packedW, Qh, Kh, Vh);
    flash_attn<<<(ROWS / 16) / 4, 128, 0, stream>>>(Qh, Kh, Vh, out);
}
